// SGAN_PoolingNet_9740985827963
// MI455X (gfx1250) — matrix operand order repaired and checked by reading, not compiled, not run on hardware
//
#include <hip/hip_runtime.h>
#include <hip/hip_bf16.h>

typedef __attribute__((ext_vector_type(16))) _Float16 v16h;
typedef __attribute__((ext_vector_type(8)))  _Float16 v8h;
typedef __attribute__((ext_vector_type(8)))  float    v8f;

#define NAGENT 1024
#define HIDDEN 64
#define KDIM   128   // 2*HIDDEN = first-layer width

union AFrag { v16h v; v8h h[2]; };

// ---------------------------------------------------------------------------
// Phase 1: fold the first layer.  A[j] = hidden[j]@W1_h + ends[j]@M + c + b1
//          B[i] = ends[i]@M,  M = W_e @ W1_e, c = b_e @ W1_e.
// Results stored as f16 (WMMA operand precision) — rounding here instead of
// in the hot loop.  1024 blocks x 128 threads, ~17 MFLOP total.
// ---------------------------------------------------------------------------
__global__ __launch_bounds__(128) void sgan_prep(
    const float* __restrict__ hidden, const float* __restrict__ track,
    const float* __restrict__ W_e,    const float* __restrict__ b_e,
    const float* __restrict__ W1,     const float* __restrict__ b1,
    _Float16* __restrict__ A, _Float16* __restrict__ Bv) {
  const int j = blockIdx.x;
  const int k = threadIdx.x;                 // output column 0..127
  const float ex = track[j * 16 + 14];       // ends[j] = group_track[j, 7, :]
  const float ey = track[j * 16 + 15];
  float m0 = 0.f, m1 = 0.f, cc = 0.f;
#pragma unroll
  for (int e = 0; e < 16; ++e) {
    const float w1v = W1[(HIDDEN + e) * KDIM + k];   // W1 rows 64..79 (embed part)
    m0 += W_e[e]      * w1v;                         // W_e[0][e]
    m1 += W_e[16 + e] * w1v;                         // W_e[1][e]
    cc += b_e[e]      * w1v;
  }
  float hh = 0.f;
  const float* hrow = hidden + j * HIDDEN;
#pragma unroll 8
  for (int h = 0; h < HIDDEN; ++h) hh += hrow[h] * W1[h * KDIM + k];
  const float bt = ex * m0 + ey * m1;
  A[j * KDIM + k]  = (_Float16)(hh + bt + cc + b1[k]);
  Bv[j * KDIM + k] = (_Float16)bt;
}

// ---------------------------------------------------------------------------
// Phase 2: pooled[i][n] = relu( max_j ( relu(A[j]-B[i]) @ W2 )[n] + b2[n] )
// One block per i (8 waves); each wave owns 8 j-blocks of 16 rows.
// GEMM tile: M=16 (j rows), N=64 (4 tiles), K=128 (4 chunks of 32) via
// v_wmma_f32_16x16x32_f16.  A operand built with packed-f16 sub+max
// (v_pk_add_f16 / v_pk_max_num_f16) that co-execute with the XDL pipe.
// ---------------------------------------------------------------------------
__global__ __launch_bounds__(256) void sgan_pool(
    const _Float16* __restrict__ A, const _Float16* __restrict__ Bv,
    const float* __restrict__ W2, const float* __restrict__ b2,
    float* __restrict__ out) {
  const int i    = blockIdx.x;
  const int wave = threadIdx.x >> 5;
  const int lane = threadIdx.x & 31;
  const int row  = lane & 15;     // A: M row in tile; B/D: N column in tile
  const int half = lane >> 4;     // K-half selector per the VGPR layouts

  // W2 fragments, resident: w2f[c][t], element e -> K = 32c + 8*half + e (+8 for e>=8),
  // N = 16t + row  (32x16 16-bit B-matrix layout).  16 x v16h = 128 VGPRs.
  v16h w2f[4][4];
#pragma unroll
  for (int c = 0; c < 4; ++c)
#pragma unroll
    for (int t = 0; t < 4; ++t)
#pragma unroll
      for (int e = 0; e < 16; ++e)
        w2f[c][t][e] = (_Float16)W2[(32 * c + 8 * half + e + (e >= 8 ? 8 : 0)) * 64 + (16 * t + row)];

  // B[i] fragment: i-constant, hoisted out of the j loop (32 VGPRs).
  const _Float16* brow = Bv + i * KDIM;
  AFrag bf[4];
#pragma unroll
  for (int c = 0; c < 4; ++c) {
    const int kb = 32 * c + 8 * half;
    bf[c].h[0] = *(const v8h*)(brow + kb);
    bf[c].h[1] = *(const v8h*)(brow + kb + 16);
  }

  float rmax[4] = {-1e30f, -1e30f, -1e30f, -1e30f};

  for (int s = 0; s < 8; ++s) {
    const int jb = wave + 8 * s;                       // this wave's j-block
    const _Float16* arow = A + (jb * 16 + row) * KDIM;
    if (s + 1 < 8)                                     // pull next A tile
      __builtin_prefetch(A + ((jb + 8) * 16 + row) * KDIM, 0, 1);

    v8f acc[4] = {};                                   // fresh per j-block
#pragma unroll
    for (int c = 0; c < 4; ++c) {
      const int kb = 32 * c + 8 * half;
      // A operand (16x32 f16 layout): elems 0..7 -> K=kb..kb+7,
      //                               elems 8..15 -> K=kb+16..kb+23
      AFrag af;
      af.h[0] = *(const v8h*)(arow + kb);
      af.h[1] = *(const v8h*)(arow + kb + 16);
      const v16h zero = {};
      const v16h y = __builtin_elementwise_max(af.v - bf[c].v, zero);  // relu, packed f16
#pragma unroll
      for (int t = 0; t < 4; ++t)
        acc[t] = __builtin_amdgcn_wmma_f32_16x16x32_f16(
            false, y, false, w2f[c][t], (short)0, acc[t], false, false);
    }
    // max over the 8 M-rows held by this lane, then running max over j-blocks
    // (relu & bias deferred: both commute with max).
#pragma unroll
    for (int t = 0; t < 4; ++t) {
      float m = acc[t][0];
#pragma unroll
      for (int r = 1; r < 8; ++r) m = fmaxf(m, acc[t][r]);
      rmax[t] = fmaxf(rmax[t], m);
    }
  }
  // merge the two M-halves (lane L <-> L+16 hold complementary rows, same N)
#pragma unroll
  for (int t = 0; t < 4; ++t)
    rmax[t] = fmaxf(rmax[t], __shfl_xor(rmax[t], 16, 32));

  __shared__ float red[8][64];
  if (lane < 16) {
#pragma unroll
    for (int t = 0; t < 4; ++t) red[wave][16 * t + row] = rmax[t];
  }
  __syncthreads();
  if (threadIdx.x < 64) {
    const int ch = threadIdx.x;
    float m = red[0][ch];
#pragma unroll
    for (int w = 1; w < 8; ++w) m = fmaxf(m, red[w][ch]);
    out[i * 64 + ch] = fmaxf(m + b2[ch], 0.f);   // final bias + relu
  }
}

// ---------------------------------------------------------------------------
extern "C" void kernel_launch(void* const* d_in, const int* in_sizes, int n_in,
                              void* d_out, int out_size, void* d_ws, size_t ws_size,
                              hipStream_t stream) {
  const float* hidden = (const float*)d_in[0];  // (1, 1024, 64)
  const float* track  = (const float*)d_in[1];  // (1024, 8, 2)
  const float* W_e    = (const float*)d_in[2];  // (2, 16)
  const float* b_e    = (const float*)d_in[3];  // (16,)
  const float* W1     = (const float*)d_in[4];  // (80, 128)
  const float* b1     = (const float*)d_in[5];  // (128,)
  const float* W2     = (const float*)d_in[6];  // (128, 64)
  const float* b2     = (const float*)d_in[7];  // (64,)
  float* out = (float*)d_out;                   // (1, 1024, 64)

  _Float16* A  = (_Float16*)d_ws;               // 1024*128 f16 = 256 KB
  _Float16* Bv = A + NAGENT * KDIM;             // 1024*128 f16 = 256 KB

  sgan_prep<<<NAGENT, 128, 0, stream>>>(hidden, track, W_e, b_e, W1, b1, A, Bv);
  sgan_pool<<<NAGENT, 256, 0, stream>>>(A, Bv, W2, b2, out);
}
